// DAAttention_59167469469720
// MI455X (gfx1250) — matrix operand order repaired and checked by reading, not compiled, not run on hardware
//
#include <hip/hip_runtime.h>

typedef __attribute__((ext_vector_type(16))) __bf16 v16bf;
typedef __attribute__((ext_vector_type(8)))  __bf16 v8bf;
typedef __attribute__((ext_vector_type(8)))  float  v8f;

#define DEV static __device__ __forceinline__

constexpr int Bsz  = 2;
constexpr int Nseq = 2048;
constexpr int Din  = 1024;
constexpr int Dout = 1024;
constexpr int NH   = 16;
constexpr int HD   = 64;           // head dim
constexpr int Mtok = Bsz * Nseq;   // 4096 tokens

// ---- helpers -------------------------------------------------------------

DEV __bf16 f2bf(float f) {
  unsigned u = __builtin_bit_cast(unsigned, f);
  u += 0x7FFFu + ((u >> 16) & 1u);               // round-to-nearest-even
  unsigned short h = (unsigned short)(u >> 16);
  return __builtin_bit_cast(__bf16, h);
}

DEV v8f zero8() {
  v8f z;
#pragma unroll
  for (int i = 0; i < 8; ++i) z[i] = 0.0f;
  return z;
}

// B-fragment (16 contiguous bf16 per lane): two b128 loads
DEV v16bf load16(const __bf16* p) {
  v8bf lo = *(const v8bf*)p;
  v8bf hi = *(const v8bf*)(p + 8);
  v16bf r;
#pragma unroll
  for (int i = 0; i < 8; ++i) { r[i] = lo[i]; r[i + 8] = hi[i]; }
  return r;
}

// A-fragment: row = lane%16; K elems colbase+{0..15} contiguous,
// colbase = kstep + (lane/16)*16 (matches contiguous B fragment).
DEV v16bf afrag(const __bf16* row, int colbase) {
  v8bf lo = *(const v8bf*)(row + colbase);
  v8bf hi = *(const v8bf*)(row + colbase + 8);
  v16bf r;
#pragma unroll
  for (int i = 0; i < 8; ++i) { r[i] = lo[i]; r[i + 8] = hi[i]; }
  return r;
}

DEV v8f wmma_bf16(v16bf a, v16bf b, v8f c) {
  return __builtin_amdgcn_wmma_f32_16x16x32_bf16(
      /*neg_a=*/false, a, /*neg_b=*/false, b,
      /*c_mod=*/(short)0, c, /*reuse_a=*/false, /*reuse_b=*/false);
}

// 16x64 tile GEMM, ping-pong software pipeline (unroll-by-2, no register
// rotation copies): while WMMAs run on one fragment set, the other set's
// loads for the next 32-wide k-slice are in flight.
DEV void gemm16x64(const __bf16* __restrict__ A, int lda, int rowbase,
                   const __bf16* __restrict__ Bt, int ldb, int colbase,
                   int K, v8f C[4], int lane) {
  const int half = lane >> 4, lr = lane & 15;
  const __bf16* arow = A + (size_t)(rowbase + lr) * lda;
  const __bf16* brow = Bt + (size_t)(colbase + lr) * ldb + half * 16;

  v16bf a0 = afrag(arow, half * 16);
  v16bf b0[4];
#pragma unroll
  for (int c = 0; c < 4; ++c) b0[c] = load16(brow + (size_t)c * 16 * ldb);

  for (int k = 0; k < K; k += 64) {       // K multiple of 64
    // prefetch set1 for k+32 while set0 computes
    v16bf a1 = afrag(arow, (k + 32) + half * 16);
    v16bf b1[4];
#pragma unroll
    for (int c = 0; c < 4; ++c) {
      const __bf16* bp = brow + (size_t)c * 16 * ldb + (k + 32);
      b1[c] = load16(bp);
      __builtin_prefetch(bp + 64, 0, 3);   // global_prefetch_b8, 2 steps ahead
    }
#pragma unroll
    for (int c = 0; c < 4; ++c) C[c] = wmma_bf16(a0, b0[c], C[c]);

    // prefetch set0 for k+64 while set1 computes
    if (k + 64 < K) {
      a0 = afrag(arow, (k + 64) + half * 16);
#pragma unroll
      for (int c = 0; c < 4; ++c)
        b0[c] = load16(brow + (size_t)c * 16 * ldb + (k + 64));
    }
#pragma unroll
    for (int c = 0; c < 4; ++c) C[c] = wmma_bf16(a1, b1[c], C[c]);
  }
}

// ---- conversion kernels --------------------------------------------------

__global__ __launch_bounds__(256) void cvt_x_kernel(const float* __restrict__ x,
                                                    __bf16* __restrict__ xb) {
  size_t i = (size_t)blockIdx.x * 256 + threadIdx.x;
  xb[i] = f2bf(x[i]);
}

// Transposed bf16 weights: dst[n][k] = W[k][n]. blockIdx.y picks matrix.
__global__ __launch_bounds__(256) void cvt_w_kernel(const float* __restrict__ Wq,
                                                    const float* __restrict__ Wk,
                                                    const float* __restrict__ Wv,
                                                    const float* __restrict__ Wo,
                                                    __bf16* __restrict__ wqkv_t,
                                                    __bf16* __restrict__ wo_t) {
  const int m = blockIdx.y;
  const float* src = (m == 0) ? Wq : (m == 1) ? Wk : (m == 2) ? Wv : Wo;
  __bf16* dst = (m < 3) ? (wqkv_t + (size_t)m * Din * Dout) : wo_t;
  size_t idx = (size_t)blockIdx.x * 256 + threadIdx.x;  // [n][k] flat
  int k = (int)(idx & 1023);
  int n = (int)(idx >> 10);
  dst[idx] = f2bf(src[(size_t)k * Dout + n]);
}

// ---- QKV projection: q,k row-major per head, v transposed per head -------

__global__ __launch_bounds__(256) void qkv_kernel(const __bf16* __restrict__ xb,
                                                  const __bf16* __restrict__ wqkv_t,
                                                  __bf16* __restrict__ qws,
                                                  __bf16* __restrict__ kws,
                                                  __bf16* __restrict__ vws) {
  const int lane = threadIdx.x & 31, half = lane >> 4, lr = lane & 15;
  const int wid = blockIdx.x * 8 + (threadIdx.x >> 5);  // 0..12287
  const int colt = wid % 48, rowt = wid / 48;
  const int rowbase = rowt * 16, colbase = colt * 64;

  v8f C[4] = {zero8(), zero8(), zero8(), zero8()};
  gemm16x64(xb, Din, rowbase, wqkv_t, Din, colbase, Din, C, lane);

  const int which = colbase >> 10;       // 0=q 1=k 2=v (uniform per wave)
  const int h = (colbase & 1023) >> 6;   // head (uniform per wave)
#pragma unroll
  for (int c = 0; c < 4; ++c) {
#pragma unroll
    for (int g = 0; g < 8; ++g) {
      const int tok = rowbase + g + 8 * half;
      const int b = tok >> 11, n = tok & 2047;
      const int d = c * 16 + lr;                     // 0..63
      const __bf16 val = f2bf(C[c][g]);
      if (which == 2)
        vws[((size_t)(b * NH + h) * HD + d) * Nseq + n] = val;      // [B,H,64,N]
      else if (which == 1)
        kws[((size_t)(b * NH + h) * Nseq + n) * HD + d] = val;      // [B,H,N,64]
      else
        qws[((size_t)(b * NH + h) * Nseq + n) * HD + d] = val;      // [B,H,N,64]
    }
  }
}

// ---- async K/V tile prefetch into LDS (CDNA5 async-to-LDS path) ----------
// K tile for a j-step is a contiguous 4KB block (K stored [N,64] row-major):
// 8 x b128 per wave.  V-transposed tile is 64 rows x 64B: 8 more.
// Flat pointers to LDS carry the LDS byte offset in addr[31:0].

DEV void async_prefetch_kv(const __bf16* Km, const __bf16* Vt, int j,
                           __bf16* bk, __bf16* bv, int lane) {
  const unsigned long long gk =
      (unsigned long long)(const char*)(Km + (size_t)j * HD);
  const unsigned long long gv =
      (unsigned long long)(const char*)Vt + (size_t)j * 2;
  const unsigned lk = (unsigned)(unsigned long long)bk;
  const unsigned lv = (unsigned)(unsigned long long)bv;
#pragma unroll
  for (int t = 0; t < 8; ++t) {              // K: contiguous 4KB
    const unsigned e = (unsigned)(t * 32 + lane) * 16u;
    asm volatile("global_load_async_to_lds_b128 %0, %1, off"
                 :: "v"(lk + e), "v"(gk + (unsigned long long)e) : "memory");
  }
#pragma unroll
  for (int t = 0; t < 8; ++t) {              // V: 64 rows (stride 4KB) x 64B
    const unsigned c = (unsigned)(t * 32 + lane);
    const unsigned d = c >> 2, r = (c & 3u) * 16u;
    asm volatile("global_load_async_to_lds_b128 %0, %1, off"
                 :: "v"(lv + d * 64u + r),
                    "v"(gv + (unsigned long long)d * (Nseq * 2) + r) : "memory");
  }
}

// ---- flash-style causal attention with distance-decay R_hat --------------

__global__ __launch_bounds__(128) void attn_kernel(const __bf16* __restrict__ qws,
                                                   const __bf16* __restrict__ kws,
                                                   const __bf16* __restrict__ vws,
                                                   const float* __restrict__ wp,
                                                   const float* __restrict__ vp,
                                                   __bf16* __restrict__ ctx) {
  __shared__ __align__(16) __bf16 ldsK[4][2][32 * HD];   // 4KB per buffer
  __shared__ __align__(16) __bf16 ldsV[4][2][HD * 32];   // 4KB per buffer
  __shared__ __align__(16) __bf16 ldsP[4][16 * 32];      // P bounce tile
  const int lane = threadIdx.x & 31, half = lane >> 4, lr = lane & 15;
  const int w = threadIdx.x >> 5;
  const int wid = blockIdx.x * 4 + w;   // 0..4095
  const int qt = wid & 127;
  const int h  = (wid >> 7) & 15;
  const int b  = wid >> 11;
  const int qbase = qt * 16;

  const __bf16* Q  = qws + (size_t)(b * NH + h) * Nseq * HD;  // [N,64]
  const __bf16* Km = kws + (size_t)(b * NH + h) * Nseq * HD;  // [N,64]
  const __bf16* Vt = vws + (size_t)(b * NH + h) * HD * Nseq;  // [64,N]
  __bf16* Pt = ldsP[w];

  const float wh = wp[h], vh = vp[h];
  const float c1 = 1.0f + __expf(vh);
  const float invs = 0.125f;   // 1/sqrt(64)

  // Q A-fragments for the two 32-wide head-dim slices (reused all j-steps)
  const __bf16* qrow = Q + (size_t)(qbase + lr) * HD;
  v16bf qa0 = afrag(qrow, 0 + half * 16);
  v16bf qa1 = afrag(qrow, 32 + half * 16);

  v8f O0 = zero8(), O1 = zero8(), O2 = zero8(), O3 = zero8();
  float rmax[8], lsum[8];
#pragma unroll
  for (int g = 0; g < 8; ++g) { rmax[g] = -1e30f; lsum[g] = 0.0f; }

  const int jend = qbase + 16;          // causal: keys [0, jend)
  async_prefetch_kv(Km, Vt, 0, ldsK[w][0], ldsV[w][0], lane);
  int cur = 0;

  for (int j = 0; j < jend; j += 32) {
    if (j + 32 < jend) {               // wave-uniform
      async_prefetch_kv(Km, Vt, j + 32, ldsK[w][cur ^ 1], ldsV[w][cur ^ 1], lane);
      asm volatile("s_wait_asynccnt 0x10" ::: "memory");  // current buf landed
    } else {
      asm volatile("s_wait_asynccnt 0x0" ::: "memory");
    }
    const __bf16* Kl = ldsK[w][cur];
    const __bf16* Vl = ldsV[w][cur];

    // ---- scores S (16x32) = Q(16x64) @ K^T(64x32): 4 WMMAs, K from LDS ----
    v8f S0 = zero8(), S1 = zero8();
    {
      const __bf16* kr = Kl + (size_t)lr * HD;          // key tokens j..j+15
      S0 = wmma_bf16(qa0, load16(kr + half * 16), S0);
      S0 = wmma_bf16(qa1, load16(kr + 32 + half * 16), S0);
    }
    {
      const __bf16* kr = Kl + (size_t)(16 + lr) * HD;   // key tokens j+16..j+31
      S1 = wmma_bf16(qa0, load16(kr + half * 16), S1);
      S1 = wmma_bf16(qa1, load16(kr + 32 + half * 16), S1);
    }

    // ---- relu * R_hat * scale, causal mask, online softmax ----
    float p0[8], p1[8], alpha[8];
#pragma unroll
    for (int g = 0; g < 8; ++g) {
      const int qi = qbase + g + 8 * half;
      const int k0 = j + lr, k1 = j + 16 + lr;
      float s0 = S0[g], s1 = S1[g];
      const float r0 = (float)(qi - k0), r1 = (float)(qi - k1);
      s0 = (s0 > 0.0f ? s0 : 0.0f) * (c1 / (1.0f + __expf(vh - wh * r0))) * invs;
      s1 = (s1 > 0.0f ? s1 : 0.0f) * (c1 / (1.0f + __expf(vh - wh * r1))) * invs;
      if (k0 > qi) s0 = -1e30f;
      if (k1 > qi) s1 = -1e30f;
      float m = fmaxf(s0, s1);
      m = fmaxf(m, __shfl_xor(m, 1, 32));
      m = fmaxf(m, __shfl_xor(m, 2, 32));
      m = fmaxf(m, __shfl_xor(m, 4, 32));
      m = fmaxf(m, __shfl_xor(m, 8, 32));   // row max within 16-lane half
      const float mn = fmaxf(rmax[g], m);
      alpha[g] = __expf(rmax[g] - mn);
      rmax[g] = mn;
      const float e0 = __expf(s0 - mn);
      const float e1 = __expf(s1 - mn);
      p0[g] = e0; p1[g] = e1;
      float rs = e0 + e1;
      rs += __shfl_xor(rs, 1, 32);
      rs += __shfl_xor(rs, 2, 32);
      rs += __shfl_xor(rs, 4, 32);
      rs += __shfl_xor(rs, 8, 32);
      lsum[g] = lsum[g] * alpha[g] + rs;
    }

    // rescale running output
#pragma unroll
    for (int g = 0; g < 8; ++g) {
      O0[g] *= alpha[g]; O1[g] *= alpha[g]; O2[g] *= alpha[g]; O3[g] *= alpha[g];
    }

    // ---- P: C-layout -> LDS -> A-layout (per-wave, s_wait_dscnt) ----
#pragma unroll
    for (int g = 0; g < 8; ++g) {
      const int row = g + 8 * half;
      Pt[row * 32 + lr]      = f2bf(p0[g]);
      Pt[row * 32 + 16 + lr] = f2bf(p1[g]);
    }
    asm volatile("s_wait_dscnt 0" ::: "memory");
    v16bf pa = afrag(Pt + lr * 32, half * 16);
    asm volatile("" ::: "memory");

    // ---- ctx (16x64) += P(16x32) @ V(32x64): 4 WMMAs, V from LDS ----
    O0 = wmma_bf16(pa, load16(Vl + (size_t)(0 * 16 + lr) * 32 + half * 16), O0);
    O1 = wmma_bf16(pa, load16(Vl + (size_t)(1 * 16 + lr) * 32 + half * 16), O1);
    O2 = wmma_bf16(pa, load16(Vl + (size_t)(2 * 16 + lr) * 32 + half * 16), O2);
    O3 = wmma_bf16(pa, load16(Vl + (size_t)(3 * 16 + lr) * 32 + half * 16), O3);

    cur ^= 1;
  }

  // ---- normalize + store ctx as bf16 [B,N,1024] ----
  __bf16* crow = ctx + (size_t)b * Nseq * Dout + (size_t)h * HD;
#pragma unroll
  for (int g = 0; g < 8; ++g) {
    const float inv = 1.0f / lsum[g];
    __bf16* p = crow + (size_t)(qbase + g + 8 * half) * Dout;
    p[0 * 16 + lr] = f2bf(O0[g] * inv);
    p[1 * 16 + lr] = f2bf(O1[g] * inv);
    p[2 * 16 + lr] = f2bf(O2[g] * inv);
    p[3 * 16 + lr] = f2bf(O3[g] * inv);
  }
}

// ---- output projection + bias, f32 out -----------------------------------

__global__ __launch_bounds__(256) void oproj_kernel(const __bf16* __restrict__ ctx,
                                                    const __bf16* __restrict__ wot,
                                                    const float* __restrict__ bo,
                                                    float* __restrict__ out) {
  const int lane = threadIdx.x & 31, half = lane >> 4, lr = lane & 15;
  const int wid = blockIdx.x * 8 + (threadIdx.x >> 5);  // 0..4095
  const int colt = wid & 15, rowt = wid >> 4;
  const int rowbase = rowt * 16, colbase = colt * 64;

  v8f C[4] = {zero8(), zero8(), zero8(), zero8()};
  gemm16x64(ctx, Dout, rowbase, wot, Dout, colbase, Dout, C, lane);

#pragma unroll
  for (int c = 0; c < 4; ++c) {
    const float bias = bo[colbase + c * 16 + lr];
#pragma unroll
    for (int g = 0; g < 8; ++g)
      out[(size_t)(rowbase + g + 8 * half) * Dout + colbase + c * 16 + lr] =
          C[c][g] + bias;
  }
}

// ---- host launch ---------------------------------------------------------

extern "C" void kernel_launch(void* const* d_in, const int* in_sizes, int n_in,
                              void* d_out, int out_size, void* d_ws, size_t ws_size,
                              hipStream_t stream) {
  (void)in_sizes; (void)n_in; (void)out_size; (void)ws_size;
  const float* x  = (const float*)d_in[0];
  const float* Wq = (const float*)d_in[1];
  const float* Wk = (const float*)d_in[2];
  const float* Wv = (const float*)d_in[3];
  const float* Wo = (const float*)d_in[4];
  const float* bo = (const float*)d_in[5];
  const float* wp = (const float*)d_in[6];
  const float* vp = (const float*)d_in[7];
  float* out = (float*)d_out;

  char* ws = (char*)d_ws;
  __bf16* xb     = (__bf16*)ws;  ws += (size_t)Mtok * Din * 2;          // 8 MiB
  __bf16* wqkv_t = (__bf16*)ws;  ws += (size_t)3 * Din * Dout * 2;      // 6 MiB
  __bf16* wo_t   = (__bf16*)ws;  ws += (size_t)Din * Dout * 2;          // 2 MiB
  __bf16* qws    = (__bf16*)ws;  ws += (size_t)Mtok * Dout * 2;         // 8 MiB
  __bf16* kws    = (__bf16*)ws;  ws += (size_t)Mtok * Dout * 2;         // 8 MiB
  __bf16* vws    = (__bf16*)ws;  ws += (size_t)Mtok * Dout * 2;         // 8 MiB
  __bf16* ctx    = (__bf16*)ws;  ws += (size_t)Mtok * Dout * 2;         // 8 MiB

  cvt_x_kernel<<<(Mtok * Din) / 256, 256, 0, stream>>>(x, xb);
  dim3 gw((Din * Dout) / 256, 4);
  cvt_w_kernel<<<gw, 256, 0, stream>>>(Wq, Wk, Wv, Wo, wqkv_t, wo_t);

  // 256 row-tiles x 48 col-tiles = 12288 waves; 8 waves/block
  qkv_kernel<<<12288 / 8, 256, 0, stream>>>(xb, wqkv_t, qws, kws, vws);

  // B*H*(N/16) = 4096 waves; 4 waves/block
  attn_kernel<<<4096 / 4, 128, 0, stream>>>(qws, kws, vws, wp, vp, ctx);

  // 256 row-tiles x 16 col-tiles = 4096 waves; 8 waves/block
  oproj_kernel<<<4096 / 8, 256, 0, stream>>>(ctx, wo_t, bo, out);
}
